// FusedAttentionWithRPB_28741921145187
// MI455X (gfx1250) — matrix operand order repaired and checked by reading, not compiled, not run on hardware
//
#include <hip/hip_runtime.h>
#include <hip/hip_bf16.h>

// ---------------- problem constants (match reference) ----------------
#define Bn 4
#define Hn 8
#define Sn 1024
#define Dn 64
#define MAXXY 30
#define MAXZ 10
#define NBXY 61
#define NTOT (Bn*Hn*Sn*Dn)   // 2,097,152

typedef __attribute__((ext_vector_type(16))) _Float16 v16h;
typedef __attribute__((ext_vector_type(8)))  _Float16 v8h;
typedef __attribute__((ext_vector_type(8)))  float    v8f;

union HV { v16h v; v8h h[2]; };

__device__ __forceinline__ v8f wmma_f16(v16h a, v16h b, v8f c) {
  return __builtin_amdgcn_wmma_f32_16x16x32_f16(false, a, false, b, (short)0, c,
                                                false, false);
}

// ---------------- preprocessing: fp32 -> fp16 (Q scaled), V transposed ----
__global__ __launch_bounds__(256) void cvt_qk_kernel(const float* __restrict__ q,
                                                     const float* __restrict__ k,
                                                     _Float16* __restrict__ qh,
                                                     _Float16* __restrict__ kh) {
  int i = blockIdx.x * 256 + threadIdx.x;
  if (i < NTOT) {
    qh[i] = (_Float16)(q[i] * 0.125f);   // fold 1/sqrt(64) into Q
    kh[i] = (_Float16)k[i];
  }
}

__global__ __launch_bounds__(256) void cvt_v_kernel(const float* __restrict__ v,
                                                    _Float16* __restrict__ vt) {
  // output layout: vt[b][h][d][s]  (d-major so PV B-frags are contiguous in s)
  int i = blockIdx.x * 256 + threadIdx.x;
  if (i < NTOT) {
    int s  = i % Sn;
    int d  = (i / Sn) % Dn;
    int bh = i / (Sn * Dn);
    vt[i] = (_Float16)v[((size_t)bh * Sn + s) * Dn + d];
  }
}

// ---------------- fused flash attention with 3D relative-position bias ----
__global__ __launch_bounds__(128) void
attn_rpb_kernel(const _Float16* __restrict__ qh, const _Float16* __restrict__ kh,
                const _Float16* __restrict__ vt,
                const int* __restrict__ px, const int* __restrict__ py,
                const int* __restrict__ pz,
                const float* __restrict__ bx, const float* __restrict__ by,
                const float* __restrict__ bz, float* __restrict__ out) {
  __shared__ float lbxy[NBXY * NBXY];                 // fused bias_x + bias_y (one head)
  __shared__ float lbz[21];
  __shared__ __align__(16) _Float16 pst[4][16 * 32];  // per-wave P staging

  const int tid  = threadIdx.x;
  const int wid  = tid >> 5;
  const int lane = tid & 31;
  const int bh   = blockIdx.x >> 4;   // 16 tile-groups per (b,h)
  const int grp  = blockIdx.x & 15;
  const int h    = bh % Hn;
  const int b    = bh / Hn;

  // build fused XY bias table + Z table for this head
  for (int i = tid; i < NBXY * NBXY; i += 128) {
    int ix = i / NBXY, iy = i - ix * NBXY;
    lbxy[i] = bx[ix * Hn + h] + by[iy * Hn + h];
  }
  for (int i = tid; i < 21; i += 128) lbz[i] = bz[i * Hn + h];
  __syncthreads();

  const int qt = grp * 4 + wid;       // query tile 0..63 (16 rows each)
  const int qb = qt * 16;
  const int hi = lane >> 4;           // lane half 0/1
  const int ln = lane & 15;

  const _Float16* Qb = qh + (size_t)(b * Hn + h) * Sn * Dn;
  const _Float16* Kb = kh + (size_t)(b * Hn + h) * Sn * Dn;
  const _Float16* Vb = vt + (size_t)(b * Hn + h) * Dn * Sn;
  const int* pxb = px + b * Sn;
  const int* pyb = py + b * Sn;
  const int* pzb = pz + b * Sn;

  // ---- Q A-frags (k order used here, same for B: halves0..15 = K 16hi..16hi+15, contiguous)
  HV aq0, aq1;
  const _Float16* qrow = Qb + (size_t)(qb + ln) * Dn;
  aq0.h[0] = *(const v8h*)(qrow + 16 * hi);
  aq0.h[1] = *(const v8h*)(qrow + 16 * hi + 8);
  aq1.h[0] = *(const v8h*)(qrow + 32 + 16 * hi);
  aq1.h[1] = *(const v8h*)(qrow + 40 + 16 * hi);

  // ---- all-ones B operand: one extra WMMA yields row-sums of P in every lane
  HV ones;
#pragma unroll
  for (int i = 0; i < 16; ++i) ones.v[i] = (_Float16)1.0f;

  // ---- per-row invariants for the 8 rows this lane-half touches
  int qxA[8], qyA[8], qzA[8], growA[8];
#pragma unroll
  for (int r = 0; r < 8; ++r) {
    int row = qb + r + 8 * hi;
    growA[r] = row;
    qxA[r] = pxb[row];
    qyA[r] = pyb[row];
    qzA[r] = pzb[row];
  }

  float mrun[8];
  v8f o[4];
  v8f lacc = (v8f){};
#pragma unroll
  for (int r = 0; r < 8; ++r) mrun[r] = -3.0e38f;
#pragma unroll
  for (int c = 0; c < 4; ++c) o[c] = (v8f){};

  _Float16* stg = &pst[wid][0];

  const int jend = qb + 15;
  for (int jb = 0; jb <= jend; jb += 32) {
    const _Float16* kr0 = Kb + (size_t)(jb + ln) * Dn;
    const _Float16* kr1 = Kb + (size_t)(jb + 16 + ln) * Dn;

    // prefetch next tile's K rows (global_prefetch_b8)
    if (jb + 32 <= jend) {
      __builtin_prefetch(kr0 + 32 * Dn, 0, 0);
      __builtin_prefetch(kr1 + 32 * Dn, 0, 0);
    }

    // ---- QK^T: S tile 16x32 as two C-frags (keys jb..jb+15, jb+16..jb+31)
    HV bk;
    v8f sl = (v8f){}, sr = (v8f){};
    bk.h[0] = *(const v8h*)(kr0 + 16 * hi);
    bk.h[1] = *(const v8h*)(kr0 + 16 * hi + 8);
    sl = wmma_f16(aq0.v, bk.v, sl);
    bk.h[0] = *(const v8h*)(kr0 + 32 + 16 * hi);
    bk.h[1] = *(const v8h*)(kr0 + 40 + 16 * hi);
    sl = wmma_f16(aq1.v, bk.v, sl);
    bk.h[0] = *(const v8h*)(kr1 + 16 * hi);
    bk.h[1] = *(const v8h*)(kr1 + 16 * hi + 8);
    sr = wmma_f16(aq0.v, bk.v, sr);
    bk.h[0] = *(const v8h*)(kr1 + 32 + 16 * hi);
    bk.h[1] = *(const v8h*)(kr1 + 40 + 16 * hi);
    sr = wmma_f16(aq1.v, bk.v, sr);

    // ---- key positions for this lane's two columns
    const int j0 = jb + ln, j1 = jb + 16 + ln;
    const int kx0 = pxb[j0], ky0 = pyb[j0], kz0 = pzb[j0];
    const int kx1 = pxb[j1], ky1 = pyb[j1], kz1 = pzb[j1];

    // ---- bias + causal mask + online softmax (rows live in 16-lane halves)
#pragma unroll
    for (int r = 0; r < 8; ++r) {
      int dx0 = qxA[r] - kx0, dy0 = qyA[r] - ky0, dz0 = qzA[r] - kz0;
      int dx1 = qxA[r] - kx1, dy1 = qyA[r] - ky1, dz1 = qzA[r] - kz1;
      dx0 = (dx0 < -MAXXY ? -MAXXY : (dx0 > MAXXY ? MAXXY : dx0)) + MAXXY;
      dy0 = (dy0 < -MAXXY ? -MAXXY : (dy0 > MAXXY ? MAXXY : dy0)) + MAXXY;
      dz0 = (dz0 < -MAXZ  ? -MAXZ  : (dz0 > MAXZ  ? MAXZ  : dz0)) + MAXZ;
      dx1 = (dx1 < -MAXXY ? -MAXXY : (dx1 > MAXXY ? MAXXY : dx1)) + MAXXY;
      dy1 = (dy1 < -MAXXY ? -MAXXY : (dy1 > MAXXY ? MAXXY : dy1)) + MAXXY;
      dz1 = (dz1 < -MAXZ  ? -MAXZ  : (dz1 > MAXZ  ? MAXZ  : dz1)) + MAXZ;
      float s0 = sl[r] + lbxy[dx0 * NBXY + dy0] + lbz[dz0];
      float s1 = sr[r] + lbxy[dx1 * NBXY + dy1] + lbz[dz1];
      if (j0 > growA[r]) s0 = -1.0e30f;
      if (j1 > growA[r]) s1 = -1.0e30f;

      float t = fmaxf(s0, s1);
      t = fmaxf(t, __shfl_xor(t, 1, 32));
      t = fmaxf(t, __shfl_xor(t, 2, 32));
      t = fmaxf(t, __shfl_xor(t, 4, 32));
      t = fmaxf(t, __shfl_xor(t, 8, 32));
      const float mn = fmaxf(mrun[r], t);
      const float corr = __expf(mrun[r] - mn);
      mrun[r] = mn;
      const float p0 = __expf(s0 - mn);
      const float p1 = __expf(s1 - mn);

      // rescale accumulators (O chunks + row-sum accumulator)
      o[0][r] *= corr; o[1][r] *= corr; o[2][r] *= corr; o[3][r] *= corr;
      lacc[r] *= corr;

      // stage P (f16) in row-major [m][n] for A-frag reload
      _Float16* pr = stg + (r + 8 * hi) * 32 + ln;
      pr[0]  = (_Float16)p0;
      pr[16] = (_Float16)p1;
    }

    // same-wave LDS RAW: drain DS counter before re-reading the staging tile
    asm volatile("s_wait_dscnt 0x0" ::: "memory");

    // ---- P as A-frag (k = local key 0..31)
    HV ap;
    const _Float16* prow = stg + ln * 32;
    ap.h[0] = *(const v8h*)(prow + 16 * hi);
    ap.h[1] = *(const v8h*)(prow + 16 * hi + 8);

    // ---- row sums of P via WMMA against ones (every lane gets its row's sum)
    lacc = wmma_f16(ap.v, ones.v, lacc);

    // ---- PV: O(16x64) += P(16x32) x V(32x64), V pre-transposed [d][s]
#pragma unroll
    for (int c = 0; c < 4; ++c) {
      HV bv;
      const _Float16* vc = Vb + (size_t)(c * 16 + ln) * Sn + jb + 16 * hi;
      bv.h[0] = *(const v8h*)(vc);
      bv.h[1] = *(const v8h*)(vc + 8);
      o[c] = wmma_f16(ap.v, bv.v, o[c]);
    }
  }

  // ---- epilogue: normalize and store fp32 output
#pragma unroll
  for (int r = 0; r < 8; ++r) {
    const float inv = 1.0f / lacc[r];
    float* orow = out + ((size_t)((b * Hn + h) * Sn + qb + r + 8 * hi)) * Dn + ln;
    orow[0]  = o[0][r] * inv;
    orow[16] = o[1][r] * inv;
    orow[32] = o[2][r] * inv;
    orow[48] = o[3][r] * inv;
  }
}

// ---------------- launcher ----------------
extern "C" void kernel_launch(void* const* d_in, const int* in_sizes, int n_in,
                              void* d_out, int out_size, void* d_ws, size_t ws_size,
                              hipStream_t stream) {
  const float* q  = (const float*)d_in[0];
  const float* k  = (const float*)d_in[1];
  const float* v  = (const float*)d_in[2];
  const int* px   = (const int*)d_in[3];
  const int* py   = (const int*)d_in[4];
  const int* pz   = (const int*)d_in[5];
  const float* bx = (const float*)d_in[6];
  const float* by = (const float*)d_in[7];
  const float* bz = (const float*)d_in[8];
  float* out = (float*)d_out;

  _Float16* qh = (_Float16*)d_ws;          // 4 MB
  _Float16* kh = qh + NTOT;                // 4 MB
  _Float16* vt = kh + NTOT;                // 4 MB (transposed V)

  cvt_qk_kernel<<<NTOT / 256, 256, 0, stream>>>(q, k, qh, kh);
  cvt_v_kernel<<<NTOT / 256, 256, 0, stream>>>(v, vt);

  // one wave32 per 16-row query tile: B*H*(S/16) = 2048 waves = 512 blocks x 4 waves
  attn_rpb_kernel<<<Bn * Hn * 16, 128, 0, stream>>>(qh, kh, vt, px, py, pz,
                                                    bx, by, bz, out);
}
